// MultiHeadAttention_72688026517754
// MI455X (gfx1250) — matrix operand order repaired and checked by reading, not compiled, not run on hardware
//
#include <hip/hip_runtime.h>
#include <stdint.h>

#define B_  4
#define S_  2048
#define D_  1024
#define H_  16
#define DK_ 64

typedef __attribute__((ext_vector_type(16))) __bf16 v16bf;
typedef __attribute__((ext_vector_type(8)))  float  v8f;
typedef __attribute__((ext_vector_type(4)))  unsigned int v4u;
typedef __attribute__((ext_vector_type(8)))  int v8i_t;
typedef __attribute__((ext_vector_type(4)))  int v4i_t;

union FragAB { v16bf v; uint32_t u[8]; };

__device__ __forceinline__ float row_max16(float x) {
  // reduce across the 16 lanes of a C-fragment row group (wave32)
  x = fmaxf(x, __shfl_xor(x, 1, 32));
  x = fmaxf(x, __shfl_xor(x, 2, 32));
  x = fmaxf(x, __shfl_xor(x, 4, 32));
  x = fmaxf(x, __shfl_xor(x, 8, 32));
  return x;
}
__device__ __forceinline__ float row_sum16(float x) {
  x += __shfl_xor(x, 1, 32);
  x += __shfl_xor(x, 2, 32);
  x += __shfl_xor(x, 4, 32);
  x += __shfl_xor(x, 8, 32);
  return x;
}

// ---------------------------------------------------------------------------
// fp32 -> bf16 conversion (weights)
// ---------------------------------------------------------------------------
__global__ void cvt_bf16_kernel(const float* __restrict__ src,
                                __bf16* __restrict__ dst, int n) {
  int i = blockIdx.x * blockDim.x + threadIdx.x;
  int stride = gridDim.x * blockDim.x;
  for (; i < n; i += stride) dst[i] = (__bf16)src[i];
}

// ---------------------------------------------------------------------------
// Y = x @ W^T + b, written head-major bf16: Yh[((b*H+h)*S+s)*DK + dk]
//   x:  [B*S, D] fp32 row-major (M = 8192, K = 1024)
//   Wb: [D, D]  bf16 row-major (torch Linear weight: row n holds W[n, :])
// Workgroup tile 128(M) x 64(N); 8 waves, each 2x2 of 16x16 WMMA tiles.
// ---------------------------------------------------------------------------
__global__ void __launch_bounds__(256)
qkv_gemm_kernel(const float* __restrict__ x, const __bf16* __restrict__ Wb,
                const float* __restrict__ bias, __bf16* __restrict__ Yh) {
  __shared__ __align__(16) __bf16 Alds[128][36];  // 72B row stride (dword ok)
  __shared__ __align__(16) __bf16 Blds[64][40];   // 80B row stride (dwordx4 ok)

  const int tid  = threadIdx.x;
  const int lane = tid & 31, w = tid >> 5;
  const int wm = w & 3, wn = w >> 2;          // 4x2 wave grid
  const int m0 = blockIdx.y * 128;
  const int n0 = blockIdx.x * 64;
  const int g  = lane >> 4, ln = lane & 15;

  v8f c[2][2] = {};

  for (int k0 = 0; k0 < D_; k0 += 32) {
    __syncthreads();
    { // A tile 128x32: fp32 load, convert to bf16 (16 elems/thread)
      int row = tid >> 1;
      int cb  = (tid & 1) * 16;
      const float4* src = (const float4*)&x[(size_t)(m0 + row) * D_ + k0 + cb];
#pragma unroll
      for (int j = 0; j < 4; ++j) {
        float4 f = src[j];
        Alds[row][cb + 4 * j + 0] = (__bf16)f.x;
        Alds[row][cb + 4 * j + 1] = (__bf16)f.y;
        Alds[row][cb + 4 * j + 2] = (__bf16)f.z;
        Alds[row][cb + 4 * j + 3] = (__bf16)f.w;
      }
    }
    { // B tile 64x32 bf16 (8 elems/thread, one b128)
      int row = tid >> 2;
      int cb  = (tid & 3) * 8;
      *(uint4*)&Blds[row][cb] =
          *(const uint4*)&Wb[(size_t)(n0 + row) * D_ + k0 + cb];
    }
    __syncthreads();

    FragAB a[2], b[2];
#pragma unroll
    for (int it = 0; it < 2; ++it)
#pragma unroll
      for (int vv = 0; vv < 8; ++vv) {
        int kk = ((vv & 3) << 1) + ((vv >> 2) << 4) + (g << 3); // A-frag layout
        a[it].u[vv] = *(const uint32_t*)&Alds[wm * 32 + it * 16 + ln][kk];
      }
#pragma unroll
    for (int jt = 0; jt < 2; ++jt)
#pragma unroll
      for (int j = 0; j < 8; ++j) {
        int k2 = ((j & 3) << 1) + ((j >> 2) << 4) + (g << 3);   // B-frag layout
        b[jt].u[j] = *(const uint32_t*)&Blds[wn * 32 + jt * 16 + ln][k2];
      }
#pragma unroll
    for (int it = 0; it < 2; ++it)
#pragma unroll
      for (int jt = 0; jt < 2; ++jt)
        c[it][jt] = __builtin_amdgcn_wmma_f32_16x16x32_bf16(
            false, a[it].v, false, b[jt].v, (short)0, c[it][jt], false, false);
  }

  // epilogue: bias add, head-major scatter, fp32 -> bf16
#pragma unroll
  for (int it = 0; it < 2; ++it)
#pragma unroll
    for (int jt = 0; jt < 2; ++jt) {
      int col = n0 + wn * 32 + jt * 16 + ln;
      float bsv = bias[col];
      int hh = col >> 6, dk = col & (DK_ - 1);
#pragma unroll
      for (int i = 0; i < 8; ++i) {
        int row = m0 + wm * 32 + it * 16 + i + 8 * g;  // C layout: M = i + 8*g
        int bb = row >> 11;          // row / S_
        int ss = row & (S_ - 1);
        float y = c[it][jt][i] + bsv;
        Yh[(((size_t)bb * H_ + hh) * S_ + ss) * DK_ + dk] = (__bf16)y;
      }
    }
}

// ---------------------------------------------------------------------------
// Flash attention, causal. One WG per (b, h, 128 query rows); 8 waves,
// each wave owns 16 query rows. Key blocks of 64 DMA'd into LDS by the
// Tensor Data Mover (TENSORcnt); V staged transposed by the threads.
// ---------------------------------------------------------------------------
__global__ void __launch_bounds__(256)
attn_kernel(const __bf16* __restrict__ Qh, const __bf16* __restrict__ Kh,
            const __bf16* __restrict__ Vh, float* __restrict__ out) {
  __shared__ __align__(16) __bf16 Klds[64][72];      // [key][dk], 144B stride
  __shared__ __align__(16) __bf16 Vt[64][72];        // [dk][key] (transposed)
  __shared__ __align__(16) __bf16 Plds[8][16][72];   // per-wave P staging

  const int tid  = threadIdx.x;
  const int lane = tid & 31, w = tid >> 5;
  const int g = lane >> 4, ln = lane & 15;
  const int q0 = blockIdx.x * 128;
  const int hh = blockIdx.y, bb = blockIdx.z;
  const size_t bh = ((size_t)bb * H_ + hh) * S_ * DK_;
  const int qr0 = q0 + w * 16;
  const float scale = 0.125f;  // 1/sqrt(DK)

  // LDS byte offset of Klds: low 32 bits of the flat LDS-aperture address
  // are the LDS offset (ISA 10.2 aperture mapping).
  const uint32_t klds_base = (uint32_t)(uintptr_t)&Klds[0][0];

  // Q rows as two 16x32 A-fragments, loaded once (register resident)
  FragAB aQ[2];
#pragma unroll
  for (int kb = 0; kb < 2; ++kb)
#pragma unroll
    for (int vv = 0; vv < 8; ++vv) {
      int kk = ((vv & 3) << 1) + ((vv >> 2) << 4) + (g << 3) + kb * 32;
      aQ[kb].u[vv] = *(const uint32_t*)&Qh[bh + (size_t)(qr0 + ln) * DK_ + kk];
    }

  float rmax[8], rsum[8];
  v8f o[4] = {};
#pragma unroll
  for (int i = 0; i < 8; ++i) { rmax[i] = -INFINITY; rsum[i] = 0.f; }

  const int kv_end = (S_ < q0 + 128) ? S_ : (q0 + 128);  // causal trim
  for (int kv0 = 0; kv0 < kv_end; kv0 += 64) {
    __syncthreads();
    if (w == 0) {
      // TDM: DMA the 64x64 bf16 K tile into Klds (rows viewed as 16 x 8B
      // elements; LDS padding 16B per 128B row reproduces Klds[64][72]).
      uint64_t ga = (uint64_t)(uintptr_t)&Kh[bh + (size_t)kv0 * DK_];
      v4u g0 = { 1u,                                   // count = 1
                 klds_base,                            // lds_addr
                 (uint32_t)ga,                         // global_addr[31:0]
                 (uint32_t)((ga >> 32) & 0x01ffffffu)  // global_addr[56:32]
                     | (2u << 30) };                   // type = 2 (image)
      v8i_t g1 = { (int)0x07130000,    // data_size=8B, pad_en, 128B/+16B pad
                   (int)(16u << 16),   // tensor_dim0 = 16 (8B units)
                   (int)(2048u << 16), // tensor_dim1 = 2048 rows
                   (int)(16u << 16),   // tile_dim0 = 16 (8B units)
                   64,                 // tile_dim1 = 64 rows
                   16,                 // tensor_dim0_stride = 16 (8B units)
                   0, 0 };
      v4i_t gz = { 0, 0, 0, 0 };
      v8i_t gz8 = { 0, 0, 0, 0, 0, 0, 0, 0 };
      __builtin_amdgcn_tensor_load_to_lds(g0, g1, gz, gz, gz8, 0);
      __builtin_amdgcn_s_wait_tensorcnt(0);
    }
    { // V staged transposed by all threads (TDM cannot transpose)
      int key = tid >> 2;
      int dkb = (tid & 3) * 16;
      const __bf16* vs = &Vh[bh + (size_t)(kv0 + key) * DK_ + dkb];
#pragma unroll
      for (int j = 0; j < 16; ++j) Vt[dkb + j][key] = vs[j];
      if (kv0 + 64 < kv_end)  // prefetch next K block (global_prefetch_b8)
        __builtin_prefetch(&Kh[bh + (size_t)(kv0 + 64 + key) * DK_ + dkb], 0, 1);
    }
    __syncthreads();

    // S = (Q K^T) over 4 key tiles of 16
    v8f s[4];
#pragma unroll
    for (int nt = 0; nt < 4; ++nt) {
      v8f acc = {};
#pragma unroll
      for (int kb = 0; kb < 2; ++kb) {
        FragAB bK;
#pragma unroll
        for (int j = 0; j < 8; ++j) {
          int k2 = ((j & 3) << 1) + ((j >> 2) << 4) + (g << 3) + kb * 32;
          bK.u[j] = *(const uint32_t*)&Klds[nt * 16 + ln][k2];
        }
        acc = __builtin_amdgcn_wmma_f32_16x16x32_bf16(
            false, aQ[kb].v, false, bK.v, (short)0, acc, false, false);
      }
      s[nt] = acc;
    }

    // scale + causal mask (element col > row -> -inf)
#pragma unroll
    for (int nt = 0; nt < 4; ++nt)
#pragma unroll
      for (int i = 0; i < 8; ++i) {
        int col = kv0 + nt * 16 + ln;
        int row = qr0 + i + 8 * g;
        float val = s[nt][i] * scale;
        s[nt][i] = (col > row) ? -INFINITY : val;
      }

    // online softmax
    float fac[8];
#pragma unroll
    for (int i = 0; i < 8; ++i) {
      float m = fmaxf(fmaxf(s[0][i], s[1][i]), fmaxf(s[2][i], s[3][i]));
      m = row_max16(m);
      float nm = fmaxf(rmax[i], m);
      fac[i] = __expf(rmax[i] - nm);
      rmax[i] = nm;
      float ps = 0.f;
#pragma unroll
      for (int nt = 0; nt < 4; ++nt) {
        float p = __expf(s[nt][i] - nm);
        s[nt][i] = p;
        ps += p;
      }
      ps = row_sum16(ps);
      rsum[i] = rsum[i] * fac[i] + ps;
    }
#pragma unroll
    for (int dt = 0; dt < 4; ++dt)
#pragma unroll
      for (int i = 0; i < 8; ++i) o[dt][i] *= fac[i];

    // restage P (C layout -> LDS row-major) for the A-fragment of PV
#pragma unroll
    for (int nt = 0; nt < 4; ++nt)
#pragma unroll
      for (int i = 0; i < 8; ++i)
        Plds[w][i + 8 * g][nt * 16 + ln] = (__bf16)s[nt][i];

    // O += P V  (same-wave LDS RAW; compiler inserts s_wait_dscnt)
#pragma unroll
    for (int kb = 0; kb < 2; ++kb) {
      FragAB aP;
#pragma unroll
      for (int vv = 0; vv < 8; ++vv) {
        int kk = ((vv & 3) << 1) + ((vv >> 2) << 4) + (g << 3) + kb * 32;
        aP.u[vv] = *(const uint32_t*)&Plds[w][ln][kk];
      }
#pragma unroll
      for (int dt = 0; dt < 4; ++dt) {
        FragAB bV;
#pragma unroll
        for (int j = 0; j < 8; ++j) {
          int k2 = ((j & 3) << 1) + ((j >> 2) << 4) + (g << 3) + kb * 32;
          bV.u[j] = *(const uint32_t*)&Vt[dt * 16 + ln][k2];
        }
        o[dt] = __builtin_amdgcn_wmma_f32_16x16x32_bf16(
            false, aP.v, false, bV.v, (short)0, o[dt], false, false);
      }
    }
  }

  // normalize and store fp32 output [B, S, D] with head interleave
#pragma unroll
  for (int i = 0; i < 8; ++i) {
    float inv = 1.0f / rsum[i];
    int row = qr0 + i + 8 * g;
#pragma unroll
    for (int dt = 0; dt < 4; ++dt) {
      int dk = dt * 16 + ln;
      out[((size_t)bb * S_ + row) * D_ + hh * DK_ + dk] = o[dt][i] * inv;
    }
  }
}

// ---------------------------------------------------------------------------
extern "C" void kernel_launch(void* const* d_in, const int* in_sizes, int n_in,
                              void* d_out, int out_size, void* d_ws, size_t ws_size,
                              hipStream_t stream) {
  // setup_inputs order: q, v, k, attn_mask, Wq, bq, Wk, bk, Wv, bv
  const float* q  = (const float*)d_in[0];
  const float* v  = (const float*)d_in[1];
  const float* k  = (const float*)d_in[2];
  const float* Wq = (const float*)d_in[4];
  const float* bq = (const float*)d_in[5];
  const float* Wk = (const float*)d_in[6];
  const float* bk = (const float*)d_in[7];
  const float* Wv = (const float*)d_in[8];
  const float* bv = (const float*)d_in[9];
  float* out = (float*)d_out;

  char* ws = (char*)d_ws;
  const size_t WB = (size_t)D_ * D_ * sizeof(__bf16);            // 2 MB each
  const size_t HB = (size_t)B_ * H_ * S_ * DK_ * sizeof(__bf16); // 16 MB each
  __bf16* Wqb = (__bf16*)(ws);
  __bf16* Wkb = (__bf16*)(ws + WB);
  __bf16* Wvb = (__bf16*)(ws + 2 * WB);
  __bf16* Qh  = (__bf16*)(ws + 3 * WB);
  __bf16* Kh  = (__bf16*)(ws + 3 * WB + HB);
  __bf16* Vh  = (__bf16*)(ws + 3 * WB + 2 * HB);

  const int nW = D_ * D_;
  cvt_bf16_kernel<<<512, 256, 0, stream>>>(Wq, Wqb, nW);
  cvt_bf16_kernel<<<512, 256, 0, stream>>>(Wk, Wkb, nW);
  cvt_bf16_kernel<<<512, 256, 0, stream>>>(Wv, Wvb, nW);

  dim3 gg(D_ / 64, (B_ * S_) / 128);
  qkv_gemm_kernel<<<gg, 256, 0, stream>>>(q, Wqb, bq, Qh);
  qkv_gemm_kernel<<<gg, 256, 0, stream>>>(k, Wkb, bk, Kh);
  qkv_gemm_kernel<<<gg, 256, 0, stream>>>(v, Wvb, bv, Vh);

  attn_kernel<<<dim3(S_ / 128, H_, B_), 256, 0, stream>>>(Qh, Kh, Vh, out);
}
